// RoomTemperatureGNNModule_59777354825872
// MI455X (gfx1250) — matrix operand order repaired and checked by reading, not compiled, not run on hardware
//
#include <hip/hip_runtime.h>
#include <hip/hip_bf16.h>

typedef __attribute__((ext_vector_type(16))) __bf16 v16bf;
typedef __attribute__((ext_vector_type(8)))  __bf16 v8bf;
typedef __attribute__((ext_vector_type(8)))  float  v8f;
typedef __attribute__((ext_vector_type(4)))  float  v4f;

#define BATCH   4
#define TSTEPS  32
#define NNODES  32
#define FIN     16
#define GHID    64
#define NEDGE   256
#define NE      288            // NEDGE + NNODES self loops
#define LSTM_IN 2048           // GHID * NNODES
#define LSTM_H  4096           // 128 * NNODES
#define G4      16384          // 4 * LSTM_H

// ---------------- workspace layout (bytes) ----------------
#define OFF_COEF   ((size_t)0)                          //  288 f32
#define OFF_SEQ    ((size_t)4096)                       //  128*2048 f32   = 1 MB
#define OFF_XPROJ  (OFF_SEQ    + (size_t)128*LSTM_IN*4) //  128*16384 f32  = 8.4 MB
#define OFF_GATES  (OFF_XPROJ  + (size_t)128*G4*4)      //  4*16384 f32
#define OFF_H      (OFF_GATES  + (size_t)BATCH*G4*4)    //  4*4096 f32
#define OFF_C      (OFF_H      + (size_t)BATCH*LSTM_H*4)
#define OFF_HBF    (OFF_C      + (size_t)BATCH*LSTM_H*4)// 4*4096 bf16
#define OFF_WHHBF  (OFF_HBF    + (size_t)BATCH*LSTM_H*2)// 16384*4096 bf16 = 134 MB

// Assemble a v16bf A-fragment from two contiguous 16-byte runs.
__device__ __forceinline__ v16bf cat8(v8bf lo, v8bf hi) {
    union { v16bf v; v8bf h[2]; } u;
    u.h[0] = lo; u.h[1] = hi;
    return u.v;
}

// ---------------- GCN edge coefficients ----------------
__global__ void k_coef(const int* __restrict__ ei, float* __restrict__ coef) {
    __shared__ float deg[NNODES];
    __shared__ float dinv[NNODES];
    int tid = threadIdx.x;
    if (tid < NNODES) deg[tid] = 0.0f;
    __syncthreads();
    if (tid < NE) {
        int dst = (tid < NEDGE) ? ei[NEDGE + tid] : (tid - NEDGE);
        atomicAdd(&deg[dst], 1.0f);
    }
    __syncthreads();
    if (tid < NNODES) {
        float d = deg[tid];
        dinv[tid] = (d > 0.0f) ? rsqrtf(fmaxf(d, 1.0f)) : 0.0f;
    }
    __syncthreads();
    if (tid < NE) {
        int src = (tid < NEDGE) ? ei[tid] : (tid - NEDGE);
        int dst = (tid < NEDGE) ? ei[NEDGE + tid] : (tid - NEDGE);
        coef[tid] = dinv[src] * dinv[dst];
    }
}

// ---------------- fused LayerNorm + GCN1 + GCN2, one block per (b,t) ----------------
__global__ void k_gcn(const float* __restrict__ x, const int* __restrict__ ei,
                      const float* __restrict__ coef,
                      const float* __restrict__ ln_g, const float* __restrict__ ln_b,
                      const float* __restrict__ W1, const float* __restrict__ b1,
                      const float* __restrict__ W2, const float* __restrict__ b2,
                      float* __restrict__ seq) {
    __shared__ float xln[NNODES * FIN];
    __shared__ float h1 [NNODES * GHID];
    __shared__ float h2 [NNODES * GHID];
    __shared__ float agg[NNODES * GHID];
    const int bt  = blockIdx.x;
    const int tid = threadIdx.x;
    const float* xp = x + (size_t)bt * NNODES * FIN;

    for (int i = tid; i < NNODES * FIN; i += blockDim.x) xln[i] = xp[i];
    __syncthreads();
    if (tid < NNODES) {                       // layernorm over F=16 per node
        float mu = 0.0f;
        for (int f = 0; f < FIN; ++f) mu += xln[tid * FIN + f];
        mu *= (1.0f / FIN);
        float var = 0.0f;
        for (int f = 0; f < FIN; ++f) { float d = xln[tid * FIN + f] - mu; var += d * d; }
        var *= (1.0f / FIN);
        float r = rsqrtf(var + 1e-5f);
        for (int f = 0; f < FIN; ++f)
            xln[tid * FIN + f] = (xln[tid * FIN + f] - mu) * r * ln_g[f] + ln_b[f];
    }
    __syncthreads();
    for (int i = tid; i < NNODES * GHID; i += blockDim.x) {   // h1 = ln(x) @ W1
        int n = i >> 6, c = i & 63;
        float s = 0.0f;
        for (int f = 0; f < FIN; ++f) s += xln[n * FIN + f] * W1[f * GHID + c];
        h1[i] = s; agg[i] = 0.0f;
    }
    __syncthreads();
    for (int idx = tid; idx < NE * GHID; idx += blockDim.x) { // aggregate 1
        int e = idx >> 6, c = idx & 63;
        int src = (e < NEDGE) ? ei[e] : (e - NEDGE);
        int dst = (e < NEDGE) ? ei[NEDGE + e] : (e - NEDGE);
        atomicAdd(&agg[dst * GHID + c], coef[e] * h1[src * GHID + c]);
    }
    __syncthreads();
    for (int i = tid; i < NNODES * GHID; i += blockDim.x) h1[i] = agg[i] + b1[i & 63];
    __syncthreads();
    for (int i = tid; i < NNODES * GHID; i += blockDim.x) {   // h2 = g1 @ W2
        int n = i >> 6, c = i & 63;
        float s = 0.0f;
        for (int f = 0; f < GHID; ++f) s += h1[n * GHID + f] * W2[f * GHID + c];
        h2[i] = s;
    }
    __syncthreads();
    for (int i = tid; i < NNODES * GHID; i += blockDim.x) agg[i] = 0.0f;
    __syncthreads();
    for (int idx = tid; idx < NE * GHID; idx += blockDim.x) { // aggregate 2
        int e = idx >> 6, c = idx & 63;
        int src = (e < NEDGE) ? ei[e] : (e - NEDGE);
        int dst = (e < NEDGE) ? ei[NEDGE + e] : (e - NEDGE);
        atomicAdd(&agg[dst * GHID + c], coef[e] * h2[src * GHID + c]);
    }
    __syncthreads();
    for (int i = tid; i < NNODES * GHID; i += blockDim.x)
        seq[(size_t)bt * LSTM_IN + i] = agg[i] + b2[i & 63];
}

// ---------------- W_hh fp32 -> bf16 (once; 134MB result is L2-resident) ----------------
__global__ void k_cvt_bf16(const float* __restrict__ w, __bf16* __restrict__ o) {
    size_t base = ((size_t)blockIdx.x * blockDim.x + threadIdx.x) * 4;
    float4 v = *(const float4*)(w + base);
    o[base + 0] = (__bf16)v.x;
    o[base + 1] = (__bf16)v.y;
    o[base + 2] = (__bf16)v.z;
    o[base + 3] = (__bf16)v.w;
}

__global__ void k_init(float* __restrict__ h, float* __restrict__ c, __bf16* __restrict__ hbf) {
    int i = blockIdx.x * blockDim.x + threadIdx.x;
    if (i < BATCH * LSTM_H) { h[i] = 0.0f; c[i] = 0.0f; hbf[i] = (__bf16)0.0f; }
}

// ---------------- x_proj = seq @ W_ih^T + b_ih + b_hh  (WMMA bf16) ----------------
// one wave per 16x16 output tile: M = bt rows (128), N = gate cols (16384)
__global__ void k_xproj(const float* __restrict__ seq, const float* __restrict__ W_ih,
                        const float* __restrict__ b_ih, const float* __restrict__ b_hh,
                        float* __restrict__ xp) {
    int wave = blockIdx.x * (blockDim.x >> 5) + (threadIdx.x >> 5);
    int lane = threadIdx.x & 31;
    int bt0 = (wave >> 10) << 4;      // 8 row tiles
    int g0  = (wave & 1023) << 4;     // 1024 col tiles
    int m   = lane & 15, hi = lane >> 4;
    const float* arow = seq + (size_t)(bt0 + m) * LSTM_IN + 8 * hi;
    v8f acc = {};
    for (int d0 = 0; d0 < LSTM_IN; d0 += 32) {
        // A frag: two contiguous 8-float runs at k = 8*hi+{0..7} and 16+8*hi+{0..7}
        v4f f0 = *(const v4f*)(arow + d0);
        v4f f1 = *(const v4f*)(arow + d0 + 4);
        v4f f2 = *(const v4f*)(arow + d0 + 16);
        v4f f3 = *(const v4f*)(arow + d0 + 20);
        v16bf a, b;
#pragma unroll
        for (int e = 0; e < 4; ++e) {
            a[e]      = (__bf16)f0[e];
            a[e + 4]  = (__bf16)f1[e];
            a[e + 8]  = (__bf16)f2[e];
            a[e + 12] = (__bf16)f3[e];
        }
#pragma unroll
        for (int e = 0; e < 16; ++e) {
            // B[k=8*hi+(e&7)+16*(e>>3)][n=lane&15] = W_ih[g0+n][d0+k]
            b[e] = (__bf16)W_ih[(size_t)(g0 + (lane & 15)) * LSTM_IN + d0 + 8 * hi + (e & 7) + ((e >> 3) << 4)];
        }
        acc = __builtin_amdgcn_wmma_f32_16x16x32_bf16(false, a, false, b,
                                                      (short)0, acc, false, false);
    }
    int n = lane & 15;
    float bias = b_ih[g0 + n] + b_hh[g0 + n];
#pragma unroll
    for (int v = 0; v < 8; ++v) {
        int bt = bt0 + v + 8 * hi;                     // C/D frag: M = v + 8*(lane/16)
        xp[(size_t)bt * G4 + g0 + n] = acc[v] + bias;
    }
}

// ---------------- per-step gates = h @ W_hh^T  (WMMA bf16, M=gates N=batch(pad16)) ----
__global__ void k_hgemm(const __bf16* __restrict__ whh, const __bf16* __restrict__ hbf,
                        float* __restrict__ gates) {
    int wave = blockIdx.x * (blockDim.x >> 5) + (threadIdx.x >> 5);
    int lane = threadIdx.x & 31;
    int g0 = wave << 4;               // 1024 tiles over 16384 gate rows
    int m  = lane & 15, hi = lane >> 4;
    const __bf16* arow = whh + (size_t)(g0 + m) * LSTM_H + 8 * hi;
    v8f acc = {};
    for (int k0 = 0; k0 < LSTM_H; k0 += 32) {
        // A frag: two 16-byte vector loads (k = 8*hi+{0..7}, 16+8*hi+{0..7})
        v8bf alo = *(const v8bf*)(arow + k0);
        v8bf ahi = *(const v8bf*)(arow + k0 + 16);
        v16bf a = cat8(alo, ahi);
        __builtin_prefetch(arow + k0 + 64, 0, 0);      // next K-chunk of W_hh
        v16bf b;
#pragma unroll
        for (int e = 0; e < 16; ++e)
            b[e] = (m < BATCH) ? hbf[m * LSTM_H + k0 + 8 * hi + (e & 7) + ((e >> 3) << 4)] : (__bf16)0.0f;
        acc = __builtin_amdgcn_wmma_f32_16x16x32_bf16(false, a, false, b,
                                                      (short)0, acc, false, false);
    }
    int n = lane & 15;                // batch index
    if (n < BATCH) {
#pragma unroll
        for (int v = 0; v < 8; ++v)
            gates[(size_t)n * G4 + g0 + v + 8 * hi] = acc[v];
    }
}

// ---------------- LSTM cell elementwise ----------------
__device__ __forceinline__ float sigf(float x) { return 1.0f / (1.0f + expf(-x)); }

__global__ void k_cell(const float* __restrict__ gates, const float* __restrict__ xp, int t,
                       float* __restrict__ c, float* __restrict__ h, __bf16* __restrict__ hbf) {
    int j = blockIdx.x * blockDim.x + threadIdx.x;     // 16384
    int b = j >> 12, u = j & 4095;
    const float* xpt = xp + (size_t)(b * TSTEPS + t) * G4;
    const float* gb  = gates + (size_t)b * G4;
    float ig = gb[u]              + xpt[u];
    float fg = gb[LSTM_H + u]     + xpt[LSTM_H + u];
    float gg = gb[2 * LSTM_H + u] + xpt[2 * LSTM_H + u];
    float og = gb[3 * LSTM_H + u] + xpt[3 * LSTM_H + u];
    float cn = sigf(fg) * c[j] + sigf(ig) * tanhf(gg);
    float hn = sigf(og) * tanhf(cn);
    c[j] = cn; h[j] = hn; hbf[j] = (__bf16)hn;
}

// ---------------- final fc: [B,N,128] @ fc_W[1,128]^T + fc_b ----------------
__global__ void k_fc(const float* __restrict__ h, const float* __restrict__ fcW,
                     const float* __restrict__ fcb, float* __restrict__ out) {
    int i = blockIdx.x * blockDim.x + threadIdx.x;     // 128 outputs
    if (i < BATCH * NNODES) {
        int b = i >> 5, n = i & 31;
        const float* hp = h + (size_t)b * LSTM_H + n * 128;
        float s = fcb[0];
        for (int k = 0; k < 128; ++k) s += hp[k] * fcW[k];
        out[i] = s;
    }
}

extern "C" void kernel_launch(void* const* d_in, const int* in_sizes, int n_in,
                              void* d_out, int out_size, void* d_ws, size_t ws_size,
                              hipStream_t stream) {
    const float* x    = (const float*)d_in[0];
    const int*   ei   = (const int*)  d_in[1];
    const float* ln_g = (const float*)d_in[2];
    const float* ln_b = (const float*)d_in[3];
    const float* W1   = (const float*)d_in[4];
    const float* b1   = (const float*)d_in[5];
    const float* W2   = (const float*)d_in[6];
    const float* b2   = (const float*)d_in[7];
    const float* W_ih = (const float*)d_in[8];
    const float* b_ih = (const float*)d_in[9];
    const float* W_hh = (const float*)d_in[10];
    const float* b_hh = (const float*)d_in[11];
    const float* fcW  = (const float*)d_in[12];
    const float* fcb  = (const float*)d_in[13];
    float* out = (float*)d_out;

    char* ws = (char*)d_ws;
    float*  coef  = (float*)(ws + OFF_COEF);
    float*  seq   = (float*)(ws + OFF_SEQ);
    float*  xproj = (float*)(ws + OFF_XPROJ);
    float*  gates = (float*)(ws + OFF_GATES);
    float*  h     = (float*)(ws + OFF_H);
    float*  c     = (float*)(ws + OFF_C);
    __bf16* hbf   = (__bf16*)(ws + OFF_HBF);
    __bf16* whhbf = (__bf16*)(ws + OFF_WHHBF);

    k_coef<<<1, 320, 0, stream>>>(ei, coef);
    k_gcn<<<BATCH * TSTEPS, 256, 0, stream>>>(x, ei, coef, ln_g, ln_b, W1, b1, W2, b2, seq);
    // 16384*4096 = 67,108,864 elems, 4 per thread
    k_cvt_bf16<<<65536, 256, 0, stream>>>(W_hh, whhbf);
    k_init<<<64, 256, 0, stream>>>(h, c, hbf);
    k_xproj<<<1024, 256, 0, stream>>>(seq, W_ih, b_ih, b_hh, xproj);   // 8192 waves
    for (int t = 0; t < TSTEPS; ++t) {
        k_hgemm<<<128, 256, 0, stream>>>(whhbf, hbf, gates);           // 1024 waves
        k_cell<<<64, 256, 0, stream>>>(gates, xproj, t, c, h, hbf);
    }
    k_fc<<<1, 128, 0, stream>>>(h, fcW, fcb, out);
    (void)in_sizes; (void)n_in; (void)out_size; (void)ws_size;
}
